// OctreeConvGn_4612794876216
// MI455X (gfx1250) — compile-verified, matrix operand order checked by reading, not run on hardware
//
#include <hip/hip_runtime.h>
#include <stdint.h>

#define CIN     64
#define COUT    128
#define KNB     27
#define NGROUP  32
#define NBATCH  8
#define CPG     4
#define KTOT    (KNB * CIN)     // 1728
#define NCHUNK  (KTOT / 32)     // 54 chunks of K=32
#define EPSF    1e-5f

typedef __attribute__((ext_vector_type(16))) __bf16 v16bf;
typedef __attribute__((ext_vector_type(8)))  float  v8f;

union Frag { uint4 u[2]; v16bf v; };   // 32 bytes: one wave32 16x32-bf16 fragment slice per lane

__device__ __forceinline__ unsigned int bfpack(__bf16 a, __bf16 b) {
    union { __bf16 h[2]; unsigned int u; } p;
    p.h[0] = a; p.h[1] = b;
    return p.u;
}

// ---------------------------------------------------------------------------
// 1) data [N, 64] f32 -> dpack [(N+1), 64 dwords]: dwords 0..31 = hi bf16 pairs,
//    dwords 32..63 = lo bf16 pairs.  Row N is all zeros (missing neighbors).
// ---------------------------------------------------------------------------
__global__ __launch_bounds__(256) void prep_data(const float* __restrict__ data,
                                                 unsigned int* __restrict__ dpack, int N) {
    int t = blockIdx.x * blockDim.x + threadIdx.x;
    int total = (N + 1) * 32;
    if (t >= total) return;
    int row = t >> 5, dp = t & 31;
    float x0 = 0.f, x1 = 0.f;
    if (row < N) {
        x0 = data[row * CIN + dp * 2 + 0];
        x1 = data[row * CIN + dp * 2 + 1];
    }
    __bf16 h0 = (__bf16)x0, h1 = (__bf16)x1;
    float  l0 = x0 - (float)h0, l1 = x1 - (float)h1;
    dpack[row * 64 + dp]      = bfpack(h0, h1);
    dpack[row * 64 + 32 + dp] = bfpack((__bf16)l0, (__bf16)l1);
}

// ---------------------------------------------------------------------------
// 2) weight [1728, 128] f32 -> WMMA B-fragment layout, hi & lo bf16 planes.
//    Layout: [ct(8)][chunk(54)][lane(32)][dword(8)]
//    lane n<16 <-> col ct*16+n holds K pairs {kb+0..7, kb+16..23},
//    lane n+16 holds {kb+8..15, kb+24..31}  (mirror of the A-matrix layout).
// ---------------------------------------------------------------------------
__global__ __launch_bounds__(256) void pack_weight(const float* __restrict__ w,
                                                   unsigned int* __restrict__ whi,
                                                   unsigned int* __restrict__ wlo) {
    int t = blockIdx.x * blockDim.x + threadIdx.x;
    if (t >= 8 * NCHUNK * 32 * 8) return;
    int d    = t & 7;
    int rem  = t >> 3;
    int lane = rem & 31; rem >>= 5;
    int chunk = rem % NCHUNK;
    int ct    = rem / NCHUNK;
    int lane16 = lane & 15, hb = lane >> 4;
    int k0  = chunk * 32 + ((d >= 4) ? 16 : 0) + hb * 8 + (d & 3) * 2;
    int col = ct * 16 + lane16;
    float w0 = w[(k0 + 0) * COUT + col];
    float w1 = w[(k0 + 1) * COUT + col];
    __bf16 h0 = (__bf16)w0, h1 = (__bf16)w1;
    float  l0 = w0 - (float)h0, l1 = w1 - (float)h1;
    whi[t] = bfpack(h0, h1);
    wlo[t] = bfpack((__bf16)l0, (__bf16)l1);
}

// ---------------------------------------------------------------------------
// 3) Fused im2col-gather + GEMM.  8 waves/block; wave = 32 rows x 128 cols
//    (2 accumulator row-tiles -> each B fragment feeds 6 WMMAs, halving the
//    redundant L2 weight traffic vs. 16-row waves).
// ---------------------------------------------------------------------------
__global__ __launch_bounds__(256) void gemm_kernel(const unsigned int* __restrict__ dpack,
                                                   const unsigned int* __restrict__ whi,
                                                   const unsigned int* __restrict__ wlo,
                                                   const int* __restrict__ neigh,
                                                   float* __restrict__ out, int N) {
    const int lane = threadIdx.x & 31;
    const int wave = threadIdx.x >> 5;
    const int base = blockIdx.x * 256 + wave * 32;   // this wave: rows [base, base+32)
    const int m    = lane & 15;
    const int hb   = lane >> 4;

    const int row0 = base + m;
    const int row1 = base + 16 + m;
    const bool v0 = row0 < N, v1 = row1 < N;
    const int rc0 = v0 ? row0 : 0;
    const int rc1 = v1 ? row1 : 0;

    v8f acc[2][8] = {};

#pragma unroll 1
    for (int j = 0; j < KNB; ++j) {
        int i0 = neigh[rc0 * KNB + j]; i0 = (v0 && i0 >= 0) ? i0 : N;   // -1 -> zero row
        int i1 = neigh[rc1 * KNB + j]; i1 = (v1 && i1 >= 0) ? i1 : N;
        const uint4* ar0 = (const uint4*)dpack + (size_t)i0 * 16;
        const uint4* ar1 = (const uint4*)dpack + (size_t)i1 * 16;
#pragma unroll
        for (int h = 0; h < 2; ++h) {                   // two CIN halves per neighbor
            Frag ah0, al0, ah1, al1;
            ah0.u[0] = ar0[h * 4 + hb];     ah0.u[1] = ar0[h * 4 + 2 + hb];
            al0.u[0] = ar0[8 + h * 4 + hb]; al0.u[1] = ar0[8 + h * 4 + 2 + hb];
            ah1.u[0] = ar1[h * 4 + hb];     ah1.u[1] = ar1[h * 4 + 2 + hb];
            al1.u[0] = ar1[8 + h * 4 + hb]; al1.u[1] = ar1[8 + h * 4 + 2 + hb];
            const int chunk = j * 2 + h;
            const uint4* bh_base = (const uint4*)(whi + ((size_t)(chunk * 32 + lane) << 3));
            const uint4* bl_base = (const uint4*)(wlo + ((size_t)(chunk * 32 + lane) << 3));
#pragma unroll
            for (int ct = 0; ct < 8; ++ct) {
                Frag bh, bl;
                bh.u[0] = bh_base[ct * 3456 + 0];       // ct stride = 54*32*8 dw = 3456 uint4
                bh.u[1] = bh_base[ct * 3456 + 1];
                bl.u[0] = bl_base[ct * 3456 + 0];
                bl.u[1] = bl_base[ct * 3456 + 1];
                // interleave the two row-tiles: independent op between dependent accumulates
                acc[0][ct] = __builtin_amdgcn_wmma_f32_16x16x32_bf16(
                    false, ah0.v, false, bh.v, (short)0, acc[0][ct], false, false);
                acc[1][ct] = __builtin_amdgcn_wmma_f32_16x16x32_bf16(
                    false, ah1.v, false, bh.v, (short)0, acc[1][ct], false, false);
                acc[0][ct] = __builtin_amdgcn_wmma_f32_16x16x32_bf16(
                    false, al0.v, false, bh.v, (short)0, acc[0][ct], false, false);
                acc[1][ct] = __builtin_amdgcn_wmma_f32_16x16x32_bf16(
                    false, al1.v, false, bh.v, (short)0, acc[1][ct], false, false);
                acc[0][ct] = __builtin_amdgcn_wmma_f32_16x16x32_bf16(
                    false, ah0.v, false, bl.v, (short)0, acc[0][ct], false, false);
                acc[1][ct] = __builtin_amdgcn_wmma_f32_16x16x32_bf16(
                    false, ah1.v, false, bl.v, (short)0, acc[1][ct], false, false);
            }
        }
    }

    // C/D layout: VGPR r -> row = tile_base + r + 8*hb, col = ct*16 + (lane&15)
#pragma unroll
    for (int tmt = 0; tmt < 2; ++tmt) {
        const int tb = base + tmt * 16 + hb * 8;
#pragma unroll
        for (int ct = 0; ct < 8; ++ct) {
#pragma unroll
            for (int r = 0; r < 8; ++r) {
                int rr = tb + r;
                if (rr < N) out[(size_t)rr * COUT + ct * 16 + m] = acc[tmt][ct][r];
            }
        }
    }
}

// ---------------------------------------------------------------------------
// 4) Group norm over sorted batch segments
// ---------------------------------------------------------------------------
__global__ void zero_stats(float* __restrict__ s) {
    int t = blockIdx.x * blockDim.x + threadIdx.x;
    if (t < 512) s[t] = 0.f;
}

__global__ void find_starts(const int* __restrict__ bid, int* __restrict__ starts, int N) {
    int b = threadIdx.x;
    if (b > NBATCH) return;
    int lo = 0, hi = N;                  // lower bound of b in sorted batch_id
    while (lo < hi) { int mid = (lo + hi) >> 1; if (bid[mid] < b) lo = mid + 1; else hi = mid; }
    starts[b] = lo;
}

__global__ __launch_bounds__(256) void stats_kernel(const float* __restrict__ x,
                                                    const int* __restrict__ starts,
                                                    float* __restrict__ ssum,
                                                    float* __restrict__ ssq) {
    __shared__ float ls[256], lq[256];
    int b  = blockIdx.y;
    int s0 = starts[b], s1 = starts[b + 1];
    int c    = threadIdx.x & 127;
    int slot = threadIdx.x >> 7;
    float sum = 0.f, sq = 0.f;
    for (int r = s0 + blockIdx.x * 2 + slot; r < s1; r += gridDim.x * 2) {
        float v = x[(size_t)r * COUT + c];
        sum += v; sq += v * v;
    }
    ls[threadIdx.x] = sum; lq[threadIdx.x] = sq;
    __syncthreads();
    if (threadIdx.x < 128) {
        ls[threadIdx.x] += ls[threadIdx.x + 128];
        lq[threadIdx.x] += lq[threadIdx.x + 128];
    }
    __syncthreads();
    if (threadIdx.x < NGROUP) {
        int g = threadIdx.x;
        float s = ls[4*g] + ls[4*g+1] + ls[4*g+2] + ls[4*g+3];
        float q = lq[4*g] + lq[4*g+1] + lq[4*g+2] + lq[4*g+3];
        atomicAdd(&ssum[b * NGROUP + g], s);
        atomicAdd(&ssq [b * NGROUP + g], q);
    }
}

__global__ void finalize_stats(const int* __restrict__ starts,
                               const float* __restrict__ ssum, const float* __restrict__ ssq,
                               float* __restrict__ mean, float* __restrict__ istd) {
    int t = threadIdx.x;                 // 256 = 8 batches * 32 groups
    int b = t >> 5;
    float cnt = (float)(starts[b + 1] - starts[b]) * (float)CPG;
    float inv = 1.0f / (cnt + EPSF);     // matches reference 1/(count+eps)
    float mu  = ssum[t] * inv;
    float var = ssq[t] * inv - mu * mu;
    mean[t] = mu;
    istd[t] = rsqrtf(var + EPSF);
}

__global__ __launch_bounds__(256) void gn_apply(float* __restrict__ out,
                                                const int* __restrict__ bid,
                                                const float* __restrict__ mean,
                                                const float* __restrict__ istd,
                                                const float* __restrict__ gnw,
                                                const float* __restrict__ gnb, int N) {
    int i = blockIdx.x * blockDim.x + threadIdx.x;
    if (i >= N * COUT) return;
    int n = i >> 7, c = i & 127;
    int s = bid[n] * NGROUP + (c >> 2);
    out[i] = (out[i] - mean[s]) * istd[s] * gnw[c] + gnb[c];
}

// ---------------------------------------------------------------------------
extern "C" void kernel_launch(void* const* d_in, const int* in_sizes, int n_in,
                              void* d_out, int out_size, void* d_ws, size_t ws_size,
                              hipStream_t stream) {
    const float* data   = (const float*)d_in[0];
    const float* weight = (const float*)d_in[1];
    const float* gnw    = (const float*)d_in[2];
    const float* gnb    = (const float*)d_in[3];
    const int*   neigh  = (const int*)d_in[4];
    const int*   bid    = (const int*)d_in[5];
    const int N = in_sizes[0] / CIN;

    unsigned int* ws = (unsigned int*)d_ws;
    size_t off = 0;
    unsigned int* dpack = ws + off; off += (size_t)(N + 1) * 64;
    unsigned int* whi   = ws + off; off += 8 * NCHUNK * 32 * 8;
    unsigned int* wlo   = ws + off; off += 8 * NCHUNK * 32 * 8;
    float* ssum  = (float*)(ws + off); off += 256;
    float* ssq   = (float*)(ws + off); off += 256;
    float* meanp = (float*)(ws + off); off += 256;
    float* istdp = (float*)(ws + off); off += 256;
    int*   starts = (int*)(ws + off);  off += 16;
    if (off * sizeof(unsigned int) > ws_size) return;

    float* conv = (float*)d_out;

    prep_data  <<<((N + 1) * 32 + 255) / 256, 256, 0, stream>>>(data, dpack, N);
    pack_weight<<<(8 * NCHUNK * 32 * 8 + 255) / 256, 256, 0, stream>>>(weight, whi, wlo);
    zero_stats <<<2, 256, 0, stream>>>(ssum);                 // ssum+ssq contiguous (512 f)
    find_starts<<<1, 32, 0, stream>>>(bid, starts, N);
    gemm_kernel<<<(N + 255) / 256, 256, 0, stream>>>(dpack, whi, wlo, neigh, conv, N);
    stats_kernel<<<dim3(64, NBATCH), 256, 0, stream>>>(conv, starts, ssum, ssq);
    finalize_stats<<<1, 256, 0, stream>>>(starts, ssum, ssq, meanp, istdp);
    gn_apply<<<(N * COUT + 255) / 256, 256, 0, stream>>>(conv, bid, meanp, istdp, gnw, gnb, N);
}
